// GNN_41824391529182
// MI455X (gfx1250) — matrix operand order repaired and checked by reading, not compiled, not run on hardware
//
#include <hip/hip_runtime.h>
#include <stdint.h>

// GCN (3-layer) fused kernel for gfx1250 (MI455X).
// One graph per workgroup (256 threads = 8 wave32). Activations live in LDS in
// f16, FEATURE-MAJOR so WMMA B-operands load as two ds_load_b128. Both GEMM
// chains (adj@x and agg@W) run on v_wmma_f32_16x16x32_f16 with f32 accumulators
// held in registers. Weight K-panels are pre-transposed to f16 in global
// scratch and streamed into LDS by the Tensor Data Mover (TENSOR_LOAD_TO_LDS),
// double-buffered one panel ahead, synchronized with s_wait_tensorcnt.

#define NGRAPH 512
#define NNODE  128
#define DIN    128
#define HID    256

// LDS pitches in halves: 2*pitch % 16 == 0 (b128 align), 2*pitch % 256 != 0
// (distinct bank groups across the 16 lanes of a half-wave).
#define APITCH 136          // sAdj [node][node]
#define XTP    136          // sXT  [feat][node] transposed activations
#define PPITCH 40           // sP   [node][K=32] agg panel (wave-local relayout)
#define WTP    40           // sWT  [out][K=32]  transposed weight panel
#define PANEL_HALVES (HID * 32)  // 8192 halves = 16 KB per weight K-panel

typedef __attribute__((ext_vector_type(16))) _Float16 v16h;
typedef __attribute__((ext_vector_type(8)))  _Float16 v8h;
typedef __attribute__((ext_vector_type(4)))  _Float16 v4h;
typedef __attribute__((ext_vector_type(8)))  float    v8f;
typedef __attribute__((ext_vector_type(4)))  unsigned int v4u;
typedef __attribute__((ext_vector_type(4)))  int      v4i;
typedef __attribute__((ext_vector_type(8)))  int      v8i;

__device__ __forceinline__ v16h cat16(v8h lo, v8h hi) {
  return __builtin_shufflevector(lo, hi, 0, 1, 2, 3, 4, 5, 6, 7,
                                         8, 9, 10, 11, 12, 13, 14, 15);
}
// A-operand: 16 contiguous halves (K = 16*lg + 0..15 for fixed M = lane%16)
__device__ __forceinline__ v16h ldsA(const _Float16* p) {
  return cat16(*(const v8h*)p, *(const v8h*)(p + 8));
}
// B-operand: 16 contiguous halves (K = 16*lg + 0..15 for fixed N = lane%16)
__device__ __forceinline__ v16h ldsB(const _Float16* p) {
  return cat16(*(const v8h*)p, *(const v8h*)(p + 8));
}
__device__ __forceinline__ v8f wmma_f16(v16h a, v16h b, v8f c) {
  return __builtin_amdgcn_wmma_f32_16x16x32_f16(false, a, false, b,
                                                (short)0, c, false, false);
}

// TDM: copy one contiguous 16 KB weight panel (8192 f16) from global scratch
// into LDS, inserting 4 DWORDs of padding after every 16 DWORDs so rows land
// at the 40-half pitch. D# per CDNA5 ISA ch.8: group0 = {count, lds_addr,
// global_addr, type=2}; group1 = {data_size=2B, pad_enable, pad_interval=3
// (16 DW), pad_amount=3 (4 DW), tensor_dim0=tile_dim0=8192, stride0=8192}.
__device__ __forceinline__ void tdm_load_panel(const _Float16* gsrc,
                                               _Float16* ldst) {
  const unsigned long long ga = (unsigned long long)(uintptr_t)gsrc;
  const unsigned lds = (unsigned)(uintptr_t)ldst;  // low 32 bits = LDS address
  v4u g0 = {1u,                                    // count=1 (valid user D#)
            lds,                                   // lds_addr [63:32]
            (unsigned)ga,                          // global_addr [95:64]
            (unsigned)((ga >> 32) & 0x01FFFFFFu) | (2u << 30)};  // [120:96]+type
  v8i g1 = {(int)((1u << 16) | (1u << 20) | (3u << 22) | (3u << 25)),
            (int)(8192u << 16),   // tensor_dim0[15:0] at bits [63:48]
            0,                    // tensor_dim0[31:16]=0, tensor_dim1[15:0]=0
            (int)(8192u << 16),   // tile_dim0 at bits [127:112]
            0,                    // tile_dim1=0 (1-D), tile_dim2=0
            8192,                 // tensor_dim0_stride[31:0]
            0, 0};
  v4i gz = {0, 0, 0, 0};
#if __clang_major__ >= 23
  v8i gz8 = {0, 0, 0, 0, 0, 0, 0, 0};
  __builtin_amdgcn_tensor_load_to_lds(g0, g1, gz, gz, gz8, 0);
#else
  __builtin_amdgcn_tensor_load_to_lds(g0, g1, gz, gz, 0);
#endif
}

// Prep: WT[p][c][k] = (f16) W[32p+k][c]; one block per 32-row K-panel,
// thread c owns output column c (32 coalesced loads, contiguous f16 stores).
__global__ void __launch_bounds__(256)
prep_wt(const float* __restrict__ W, _Float16* __restrict__ dst) {
  const int p = blockIdx.x;
  const int c = threadIdx.x;
  const float* wg = W + (size_t)(p * 32) * HID + c;
  _Float16* o = dst + (size_t)p * PANEL_HALVES + c * 32;
#pragma unroll
  for (int q = 0; q < 4; ++q) {
    v8h h;
#pragma unroll
    for (int i = 0; i < 8; ++i)
      h[i] = (_Float16)wg[(size_t)(q * 8 + i) * HID];
    *(v8h*)(o + q * 8) = h;
  }
}

// One GCN layer: sXT <- relu((adj @ X) @ W + b)^T over 32-wide K panels.
// Wave m owns node rows [16m,16m+16); its agg panel in sP is exactly the
// A-operand it consumes (wave-local LDS, in-order, no barrier needed).
template <bool TDM>
__device__ __forceinline__ void gcn_layer(
    const _Float16* __restrict__ sAdj, _Float16* __restrict__ sXT,
    _Float16* __restrict__ sP, _Float16* __restrict__ sW0,
    _Float16* __restrict__ sW1, const float* __restrict__ W,
    const _Float16* __restrict__ wt, const float* __restrict__ Bv,
    int numPanels, int tid) {
  const int lane = tid & 31;
  const int wave = tid >> 5;
  const int lg   = lane >> 4;
  const int lm   = lane & 15;
  const int row0 = wave * 16;

  v8f acc[16] = {};  // 16 output tiles (16x16 f32) per wave

  if (TDM && wave == 0) tdm_load_panel(wt, sW0);  // prefetch panel 0

  for (int p = 0; p < numPanels; ++p) {
    _Float16* sWT = (p & 1) ? sW1 : sW0;

    if (TDM) {
      if (wave == 0) {  // issue next panel, retire current (TDM is in-order)
        if (p + 1 < numPanels) {
          tdm_load_panel(wt + (size_t)(p + 1) * PANEL_HALVES,
                         (p & 1) ? sW0 : sW1);
          __builtin_amdgcn_s_wait_tensorcnt((short)1);
        } else {
          __builtin_amdgcn_s_wait_tensorcnt((short)0);
        }
      }
      __syncthreads();  // panel p visible to all waves
    } else {
      // fallback: stage W[32p..32p+32) transposed by hand (coalesced loads)
      const float* wg = W + (size_t)(p * 32) * HID + tid;
#pragma unroll
      for (int q = 0; q < 4; ++q) {
        v8h h;
#pragma unroll
        for (int i = 0; i < 8; ++i)
          h[i] = (_Float16)wg[(size_t)(q * 8 + i) * HID];
        *(v8h*)(sWT + tid * WTP + q * 8) = h;
      }
      __syncthreads();
    }

    // ---- aggregation: sP(own rows) = (adj @ X)[row0..row0+16, 32p..32p+32)
    v8f p0 = {}, p1 = {};
#pragma unroll
    for (int k = 0; k < 4; ++k) {  // K over 128 nodes
      v16h a  = ldsA(sAdj + (row0 + lm) * APITCH + k * 32 + 16 * lg);
      v16h b0 = ldsB(sXT + (p * 32 + lm) * XTP + k * 32 + 16 * lg);
      v16h b1 = ldsB(sXT + (p * 32 + 16 + lm) * XTP + k * 32 + 16 * lg);
      p0 = wmma_f16(a, b0, p0);
      p1 = wmma_f16(a, b1, p1);
    }
    // D layout: VGPR r = row (row0 + r + 8*lg), col = lm. Wave-local spill.
#pragma unroll
    for (int r = 0; r < 8; ++r) {
      sP[(row0 + 8 * lg + r) * PPITCH + lm]      = (_Float16)p0[r];
      sP[(row0 + 8 * lg + r) * PPITCH + 16 + lm] = (_Float16)p1[r];
    }

    // ---- dense: acc[n] += panel(16x32, own rows) x Wpanel(32 x 16n)
    v16h a = ldsA(sP + (row0 + lm) * PPITCH + 16 * lg);
#pragma unroll
    for (int n = 0; n < 16; ++n) {
      v16h b = ldsB(sWT + (n * 16 + lm) * WTP + 16 * lg);
      acc[n] = wmma_f16(a, b, acc[n]);
    }

    if (TDM) __syncthreads();  // all reads of sWT done before DMA reuses it
  }

  __syncthreads();  // all waves' agg reads of sXT done before overwrite

  // ---- epilogue: bias + ReLU -> sXT (transposed); 8 consecutive rows of one
  // column are contiguous -> one ds_store_b128 per tile-column.
#pragma unroll
  for (int n = 0; n < 16; ++n) {
    const float bias = Bv[n * 16 + lm];
    v8h h;
#pragma unroll
    for (int r = 0; r < 8; ++r) {
      float v = acc[n][r] + bias;
      h[r] = (_Float16)(v > 0.f ? v : 0.f);
    }
    *(v8h*)(sXT + (n * 16 + lm) * XTP + row0 + 8 * lg) = h;
  }
  __syncthreads();
}

template <bool TDM>
__global__ void __launch_bounds__(256, 1)
gcn3_fused(const float* __restrict__ xin, const float* __restrict__ adjg,
           const float* __restrict__ W1, const float* __restrict__ B1,
           const float* __restrict__ W2, const float* __restrict__ B2,
           const float* __restrict__ W3, const float* __restrict__ B3,
           const float* __restrict__ Wh, const float* __restrict__ bh,
           const _Float16* __restrict__ wt1, const _Float16* __restrict__ wt2,
           const _Float16* __restrict__ wt3, float* __restrict__ out) {
  __shared__ _Float16 sAdj[NNODE * APITCH];  // 34,816 B
  __shared__ _Float16 sXT[HID * XTP];        // 69,632 B (feature-major)
  __shared__ _Float16 sP[NNODE * PPITCH];    // 10,240 B
  __shared__ _Float16 sW0[HID * WTP];        // 20,480 B
  __shared__ _Float16 sW1[HID * WTP];        // 20,480 B  (total 155,648 B)

  const int g   = blockIdx.x;
  const int tid = threadIdx.x;

  // ---- stage adjacency row-major (A-operand side), f32 -> f16 ----
  {
    const float4* src = (const float4*)(adjg + (size_t)g * NNODE * NNODE);
    for (int i = tid; i < NNODE * (NNODE / 4); i += 256) {  // 16 iters, uniform
      const int r = i >> 5;
      const int c = (i & 31) << 2;
      float4 v = src[i];
      v4h h = {(_Float16)v.x, (_Float16)v.y, (_Float16)v.z, (_Float16)v.w};
      *(v4h*)(sAdj + r * APITCH + c) = h;
    }
  }
  // ---- stage input features transposed: sXT[c][r] = x[r][c] (coalesced) ----
  {
    const int c  = tid & 127;
    const int r0 = (tid >> 7) * 64;
    const float* xg = xin + (size_t)g * NNODE * DIN + c;
#pragma unroll
    for (int q = 0; q < 8; ++q) {
      v8h h;
#pragma unroll
      for (int i = 0; i < 8; ++i)
        h[i] = (_Float16)xg[(size_t)(r0 + q * 8 + i) * DIN];
      *(v8h*)(sXT + c * XTP + r0 + q * 8) = h;
    }
  }
  __syncthreads();

  // ---- three GCN layers (layer 1: K=128 -> 4 panels; layers 2-3: 8) ----
  gcn_layer<TDM>(sAdj, sXT, sP, sW0, sW1, W1, wt1, B1, DIN / 32, tid);
  gcn_layer<TDM>(sAdj, sXT, sP, sW0, sW1, W2, wt2, B2, HID / 32, tid);
  gcn_layer<TDM>(sAdj, sXT, sP, sW0, sW1, W3, wt3, B3, HID / 32, tid);

  // ---- head: mean over nodes, dot with Wh, + bh; columns contiguous ----
  float s = 0.f;
  {
    const _Float16* col = sXT + tid * XTP;  // feature f = tid (HID == 256)
#pragma unroll
    for (int q = 0; q < 16; ++q) {
      v8h h = *(const v8h*)(col + q * 8);
#pragma unroll
      for (int i = 0; i < 8; ++i) s += (float)h[i];
    }
  }
  s *= Wh[tid];

  float* red = (float*)sP;  // sP dead; reuse as f32 reduction scratch
  red[tid] = s;
  __syncthreads();
#pragma unroll
  for (int off = 128; off > 0; off >>= 1) {
    if (tid < off) red[tid] += red[tid + off];
    __syncthreads();
  }
  if (tid == 0) out[g] = red[0] * (1.f / (float)NNODE) + bh[0];
}

extern "C" void kernel_launch(void* const* d_in, const int* in_sizes, int n_in,
                              void* d_out, int out_size, void* d_ws, size_t ws_size,
                              hipStream_t stream) {
  const float* xin = (const float*)d_in[0];
  const float* adj = (const float*)d_in[1];
  const float* W1  = (const float*)d_in[2];
  const float* B1  = (const float*)d_in[3];
  const float* W2  = (const float*)d_in[4];
  const float* B2  = (const float*)d_in[5];
  const float* W3  = (const float*)d_in[6];
  const float* B3  = (const float*)d_in[7];
  const float* Wh  = (const float*)d_in[8];
  const float* bh  = (const float*)d_in[9];
  float* out = (float*)d_out;

  const size_t wtBytes = (size_t)(4 + 8 + 8) * PANEL_HALVES * sizeof(_Float16);
  _Float16* wt1 = (_Float16*)d_ws;
  _Float16* wt2 = wt1 + (size_t)4 * PANEL_HALVES;
  _Float16* wt3 = wt2 + (size_t)8 * PANEL_HALVES;

  if (ws_size >= wtBytes) {
    // Pre-transpose weights to f16 panel-major scratch, then TDM-fed kernel.
    prep_wt<<<dim3(4), dim3(256), 0, stream>>>(W1, wt1);
    prep_wt<<<dim3(8), dim3(256), 0, stream>>>(W2, wt2);
    prep_wt<<<dim3(8), dim3(256), 0, stream>>>(W3, wt3);
    gcn3_fused<true><<<dim3(NGRAPH), dim3(256), 0, stream>>>(
        xin, adj, W1, B1, W2, B2, W3, B3, Wh, bh, wt1, wt2, wt3, out);
  } else {
    gcn3_fused<false><<<dim3(NGRAPH), dim3(256), 0, stream>>>(
        xin, adj, W1, B1, W2, B2, W3, B3, Wh, bh, wt1, wt2, wt3, out);
  }
}
